// CrossAttention_5291399709226
// MI455X (gfx1250) — matrix operand order repaired and checked by reading, not compiled, not run on hardware
//
#include <hip/hip_runtime.h>
#include <stdint.h>

#define B_   4
#define SQ   2048
#define SKV  2048
#define DM   1024
#define NH   16
#define DH   64

typedef __attribute__((ext_vector_type(16))) __bf16 v16bf;
typedef __attribute__((ext_vector_type(8)))  float  v8f;
typedef __attribute__((ext_vector_type(4)))  int    v4i;

#if __has_builtin(__builtin_amdgcn_global_load_async_to_lds_b128)
#define HAVE_ASYNC_LDS 1
#else
#define HAVE_ASYNC_LDS 0
#endif

__device__ __forceinline__ v8f wmma_bf16(v16bf a, v16bf b, v8f c) {
  // D = A(16x32 bf16) * B(32x16 bf16) + C(16x16 f32)
  return __builtin_amdgcn_wmma_f32_16x16x32_bf16(
      /*neg_a=*/false, a, /*neg_b=*/false, b,
      /*c_mod=*/(short)0, c, /*reuse_a=*/false, /*reuse_b=*/false);
}

union Frag { v16bf v; uint32_t u[8]; };

// A fragment, 16x32 bf16, row-major source. Ap points at (row0, k0), lda even.
__device__ __forceinline__ v16bf load_a_rm(const __bf16* Ap, int lda) {
  const int l = threadIdx.x & 31, m = l & 15, hl = l >> 4;
  const uint32_t* row = (const uint32_t*)(Ap + (size_t)m * lda);
  Frag f;
#pragma unroll
  for (int i = 0; i < 8; ++i) {
    const int kb = (i < 4) ? (hl * 8 + 2 * i) : (16 + hl * 8 + 2 * (i - 4));
    f.u[i] = row[kb >> 1];
  }
  return f.v;
}

// B fragment (32x16) from K-contiguous storage Bt[n][k]: lane = column n,
// half h lanes hold K=8h..8h+7 and K=16+8h..16+8h+7 (same order as A).
__device__ __forceinline__ v16bf load_b_tr(const __bf16* Bt, int ldt) {
  const int l = threadIdx.x & 31, n = l & 15, hl = l >> 4;
  const uint32_t* row = (const uint32_t*)(Bt + (size_t)n * ldt);
  Frag f;
#pragma unroll
  for (int j = 0; j < 8; ++j) f.u[j] = row[(j < 4) ? (hl * 4 + j) : (8 + hl * 4 + (j - 4))];
  return f.v;
}

// 16-lane max reduction, VALU-only via DPP row rotations (rows = 16 lanes).
__device__ __forceinline__ float rowmax16(float v) {
#if __has_builtin(__builtin_amdgcn_update_dpp)
  union { float f; int i; } a, b;
  a.f = v;
  b.i = __builtin_amdgcn_update_dpp(0, a.i, 0x128, 0xF, 0xF, false); // row_ror:8
  a.f = fmaxf(a.f, b.f);
  b.i = __builtin_amdgcn_update_dpp(0, a.i, 0x124, 0xF, 0xF, false); // row_ror:4
  a.f = fmaxf(a.f, b.f);
  b.i = __builtin_amdgcn_update_dpp(0, a.i, 0x122, 0xF, 0xF, false); // row_ror:2
  a.f = fmaxf(a.f, b.f);
  b.i = __builtin_amdgcn_update_dpp(0, a.i, 0x121, 0xF, 0xF, false); // row_ror:1
  a.f = fmaxf(a.f, b.f);
  return a.f;
#else
#pragma unroll
  for (int d = 8; d >= 1; d >>= 1) v = fmaxf(v, __shfl_xor(v, d, 32));
  return v;
#endif
}

// 16B global -> LDS copy: async (ASYNCcnt-tracked) when available.
// Builtin signature (probed via diagnostic): (v4i AS1*, v4i AS3*, imm, imm).
// Integer detours are exact on AMDGPU: AS1 == flat VA for global memory;
// AS3 offset == low 32 bits of the flat LDS address (ISA §10.2 aperture rules).
__device__ __forceinline__ void copy16_to_lds(__bf16* dst_lds, const __bf16* src_glb) {
#if HAVE_ASYNC_LDS
  typedef __attribute__((address_space(1))) v4i glb_v4i;
  typedef __attribute__((address_space(3))) v4i lds_v4i;
  __builtin_amdgcn_global_load_async_to_lds_b128(
      (glb_v4i*)(uintptr_t)src_glb,
      (lds_v4i*)(uint32_t)(uintptr_t)dst_lds, 0, 0);
#else
  *(uint4*)dst_lds = *(const uint4*)src_glb;
#endif
}

__device__ __forceinline__ void wait_async0() {
#if HAVE_ASYNC_LDS
#if __has_builtin(__builtin_amdgcn_s_wait_asynccnt)
  __builtin_amdgcn_s_wait_asynccnt(0);
#else
  asm volatile("s_wait_asynccnt 0" ::: "memory");
#endif
#endif
}

// ---------------- f32 -> bf16 conversion (vectorized x4) ----------------
__global__ __launch_bounds__(256) void cvt_bf16_kernel(
    const float* __restrict__ in, __bf16* __restrict__ out, int n4) {
  const int i = blockIdx.x * blockDim.x + threadIdx.x;
  if (i < n4) {
    const float4 f = ((const float4*)in)[i];
    union { __bf16 h[4]; uint2 u; } r;
    r.h[0] = (__bf16)f.x; r.h[1] = (__bf16)f.y;
    r.h[2] = (__bf16)f.z; r.h[3] = (__bf16)f.w;
    ((uint2*)out)[i] = r.u;
  }
}

// f32 [K][N] -> bf16 [N][K] transpose-convert; batched over blockIdx.y.
__global__ __launch_bounds__(256) void cvt_tr_kernel(
    const float* __restrict__ in, __bf16* __restrict__ out, int K, int N) {
  in  += (size_t)blockIdx.y * K * N;
  out += (size_t)blockIdx.y * K * N;
  const int idx = blockIdx.x * blockDim.x + threadIdx.x;
  if (idx < K * N) {
    const int n = idx / K, k = idx - n * K;
    out[idx] = (__bf16)in[(size_t)k * N + n];
  }
}

// ---------------- generic batched bf16 WMMA GEMM ----------------
// C[M,N] = A[M,K] @ B[K,N] + bias[N], with B pre-transposed: Bt[N][K] bf16.
// MODE 0: f32 row-major out; 1: bf16 row-major out; 2: bf16 transposed out
template <int MODE>
__global__ __launch_bounds__(128) void gemm_bf16_kernel(
    const __bf16* __restrict__ A, const __bf16* __restrict__ Bt,
    const float* __restrict__ bias, void* __restrict__ Cout,
    int M, int N, int K, int inner,
    size_t sA, size_t sB, size_t sBias, size_t sC) {
  const int z = blockIdx.z;
  A    += (size_t)(z / inner) * sA;
  Bt   += (size_t)(z % inner) * sB;
  bias += (size_t)(z % inner) * sBias;

  const int wv = threadIdx.x >> 5;
  const int m0 = blockIdx.y * 64 + wv * 16;   // wave owns 16 rows
  const int n0 = blockIdx.x * 64;             // 4 column tiles

  v8f acc[4] = {};
  for (int k0 = 0; k0 < K; k0 += 32) {
    const v16bf a = load_a_rm(A + (size_t)m0 * K + k0, K);
#pragma unroll
    for (int t = 0; t < 4; ++t) {
      const v16bf b = load_b_tr(Bt + (size_t)(n0 + t * 16) * K + k0, K);
      acc[t] = wmma_bf16(a, b, acc[t]);
    }
  }

  const int l = threadIdx.x & 31, n = l & 15, hl = l >> 4;
#pragma unroll
  for (int t = 0; t < 4; ++t) {
#pragma unroll
    for (int r = 0; r < 8; ++r) {
      const int row = m0 + r + 8 * hl;
      const int col = n0 + t * 16 + n;
      const float vres = acc[t][r] + bias[col];
      if (MODE == 0)
        ((float*)Cout)[(size_t)z * sC + (size_t)row * N + col] = vres;
      else if (MODE == 1)
        ((__bf16*)Cout)[(size_t)z * sC + (size_t)row * N + col] = (__bf16)vres;
      else
        ((__bf16*)Cout)[(size_t)z * sC + (size_t)col * M + row] = (__bf16)vres;
    }
  }
}

// ---------------- flash attention ----------------
// grid: (SQ/64, B*NH), block 128 (4 waves). Wave owns 16 query rows, KV chunk 64.
// K/V tiles are shared by all 4 waves -> staged once per WG into double-buffered
// LDS via async global->LDS copies; fragments then come from DS loads.
__global__ __launch_bounds__(128) void flash_attn_kernel(
    const __bf16* __restrict__ Q, const __bf16* __restrict__ Kc,
    const __bf16* __restrict__ Vt, __bf16* __restrict__ heads) {
  const int bh = blockIdx.y, h = bh % NH, b = bh / NH;
  const int wv = threadIdx.x >> 5;
  const int tid = threadIdx.x;
  const int q0 = blockIdx.x * 64 + wv * 16;

  const __bf16* Qp = Q  + (size_t)bh * SQ  * DH;
  const __bf16* Kp = Kc + (size_t)bh * SKV * DH;
  const __bf16* Vp = Vt + (size_t)bh * DH  * SKV;

  __shared__ alignas(16) __bf16 ldsK[2][64 * 64];  // [kv_local][d]
  __shared__ alignas(16) __bf16 ldsV[2][64 * 64];  // [d][kv_local]
  __shared__ alignas(16) __bf16 ldsP[4][16 * 64];  // per-wave P staging
  __bf16* myP = &ldsP[wv][0];

  // stage one 64-wide KV chunk into LDS buffer `buf` (128 threads cooperative)
  auto stage = [&](int buf, int kv0) {
    const __bf16* ksrc = Kp + (size_t)kv0 * DH;    // contiguous 8KB block
#pragma unroll
    for (int i = 0; i < 4; ++i) {
      const int idx = tid + i * 128;               // 0..511 16B transfers
      copy16_to_lds(&ldsK[buf][idx * 8], ksrc + (size_t)idx * 8);
      const int row = idx >> 3, c8 = idx & 7;      // V: 64 rows x 128B
      copy16_to_lds(&ldsV[buf][row * 64 + c8 * 8],
                    Vp + (size_t)row * SKV + kv0 + c8 * 8);
    }
  };

  // Q fragments stay in registers for the whole KV sweep
  const v16bf qa0 = load_a_rm(Qp + (size_t)q0 * DH + 0,  DH);
  const v16bf qa1 = load_a_rm(Qp + (size_t)q0 * DH + 32, DH);

  // all-ones B fragment for row-sum accumulation (bf16 1.0 = 0x3F80)
  Frag onesf;
#pragma unroll
  for (int j = 0; j < 8; ++j) onesf.u[j] = 0x3F803F80u;
  const v16bf ones = onesf.v;

  const int l = threadIdx.x & 31, n = l & 15, hl = l >> 4;

  float mrow[8];
#pragma unroll
  for (int r = 0; r < 8; ++r) mrow[r] = -1e30f;
  v8f o[4] = {};
  v8f lacc = {};                                   // running row sums

  stage(0, 0);                                     // prologue prefetch

  const int NCH = SKV / 64;
  for (int c = 0; c < NCH; ++c) {
    wait_async0();           // own async issues (chunk c) complete
    __syncthreads();         // chunk c visible; all waves done reading buf (c+1)&1
    if (c + 1 < NCH) stage((c + 1) & 1, (c + 1) * 64);

    const __bf16* Kt = &ldsK[c & 1][0];
    const __bf16* Vl = &ldsV[c & 1][0];

    // S = Q @ K^T from LDS (contiguous d per kv row)
    v8f s[4];
#pragma unroll
    for (int t = 0; t < 4; ++t) {
      v8f zz = {};
      zz = wmma_bf16(qa0, load_b_tr(Kt + (size_t)(t * 16) * 64 + 0,  64), zz);
      zz = wmma_bf16(qa1, load_b_tr(Kt + (size_t)(t * 16) * 64 + 32, 64), zz);
      s[t] = zz;
    }

    float alpha[8];
#pragma unroll
    for (int r = 0; r < 8; ++r) {
      const float v0 = s[0][r] * 0.125f;          // 1/sqrt(64)
      const float v1 = s[1][r] * 0.125f;
      const float v2 = s[2][r] * 0.125f;
      const float v3 = s[3][r] * 0.125f;
      const float mx = rowmax16(fmaxf(fmaxf(v0, v1), fmaxf(v2, v3)));
      const float mnew = fmaxf(mrow[r], mx);
      alpha[r] = __expf(mrow[r] - mnew);
      mrow[r] = mnew;
      const float p0 = __expf(v0 - mnew);
      const float p1 = __expf(v1 - mnew);
      const float p2 = __expf(v2 - mnew);
      const float p3 = __expf(v3 - mnew);
      __bf16* rowp = myP + (size_t)(r + 8 * hl) * 64;
      rowp[n]      = (__bf16)p0;
      rowp[16 + n] = (__bf16)p1;
      rowp[32 + n] = (__bf16)p2;
      rowp[48 + n] = (__bf16)p3;
    }

    const v16bf pa0 = load_a_rm(myP,      64);    // P columns kv 0..31 (DS in-order)
    const v16bf pa1 = load_a_rm(myP + 32, 64);    // P columns kv 32..63

    // row-sum accumulator: lacc = lacc*alpha + P @ ones
#pragma unroll
    for (int r = 0; r < 8; ++r) lacc[r] = lacc[r] * alpha[r];
    lacc = wmma_bf16(pa0, ones, lacc);
    lacc = wmma_bf16(pa1, ones, lacc);

    // O = O*alpha + P @ V from LDS (V tile is [d][kv_local])
#pragma unroll
    for (int t = 0; t < 4; ++t) {
#pragma unroll
      for (int r = 0; r < 8; ++r) o[t][r] = o[t][r] * alpha[r];
      o[t] = wmma_bf16(pa0, load_b_tr(Vl + (size_t)(t * 16) * 64 + 0,  64), o[t]);
      o[t] = wmma_bf16(pa1, load_b_tr(Vl + (size_t)(t * 16) * 64 + 32, 64), o[t]);
    }
  }

  // epilogue: normalize by row sums, write concat-heads row-major bf16 [B*SQ][NH*DH]
  __bf16* out = heads + ((size_t)b * SQ) * (NH * DH) + (size_t)h * DH;
#pragma unroll
  for (int t = 0; t < 4; ++t) {
#pragma unroll
    for (int r = 0; r < 8; ++r) {
      const int row = q0 + r + 8 * hl;
      const int col = t * 16 + n;
      out[(size_t)row * (NH * DH) + col] = (__bf16)(o[t][r] / lacc[r]);
    }
  }
}

extern "C" void kernel_launch(void* const* d_in, const int* in_sizes, int n_in,
                              void* d_out, int out_size, void* d_ws, size_t ws_size,
                              hipStream_t stream) {
  (void)in_sizes; (void)n_in; (void)out_size; (void)ws_size;
  const float* x   = (const float*)d_in[0];
  const float* enc = (const float*)d_in[1];
  const float* Wq  = (const float*)d_in[2];
  const float* bq  = (const float*)d_in[3];
  const float* Wk  = (const float*)d_in[4];
  const float* bk  = (const float*)d_in[5];
  const float* Wv  = (const float*)d_in[6];
  const float* bv  = (const float*)d_in[7];
  const float* Wo  = (const float*)d_in[8];
  const float* bo  = (const float*)d_in[9];
  float* out = (float*)d_out;

  // carve bf16 workspace (~105 MB total)
  uintptr_t w = (uintptr_t)d_ws;
  auto carve = [&](size_t elems) {
    __bf16* p = (__bf16*)w;
    w += (elems * 2 + 255) & ~(size_t)255;
    return p;
  };
  const size_t NX = (size_t)B_ * SQ * DM;
  const size_t NW = (size_t)NH * DM * DH;
  const size_t NQ = (size_t)B_ * NH * SQ * DH;
  __bf16* xb   = carve(NX);
  __bf16* eb   = carve(NX);
  __bf16* wqtb = carve(NW);                     // [h][n][k] transposed
  __bf16* wktb = carve(NW);
  __bf16* wvtb = carve(NW);
  __bf16* wotb = carve((size_t)DM * DM);        // [n][k] transposed
  __bf16* Qb   = carve(NQ);
  __bf16* Kb   = carve(NQ);
  __bf16* Vtb  = carve(NQ);
  __bf16* Hb   = carve((size_t)B_ * SQ * NH * DH);

  {
    const int n4 = (int)(NX / 4);
    cvt_bf16_kernel<<<dim3((n4 + 255) / 256), dim3(256), 0, stream>>>(x, xb, n4);
    cvt_bf16_kernel<<<dim3((n4 + 255) / 256), dim3(256), 0, stream>>>(enc, eb, n4);
  }
  {
    const dim3 g((DM * DH + 255) / 256, NH);
    cvt_tr_kernel<<<g, dim3(256), 0, stream>>>(Wq, wqtb, DM, DH);
    cvt_tr_kernel<<<g, dim3(256), 0, stream>>>(Wk, wktb, DM, DH);
    cvt_tr_kernel<<<g, dim3(256), 0, stream>>>(Wv, wvtb, DM, DH);
    const dim3 go((DM * DM + 255) / 256, 1);
    cvt_tr_kernel<<<go, dim3(256), 0, stream>>>(Wo, wotb, DM, DM);
  }

  const dim3 blk(128);
  const dim3 gp(1, SQ / 64, B_ * NH);
  gemm_bf16_kernel<1><<<gp, blk, 0, stream>>>(xb, wqtb, bq, Qb, SQ, DH, DM, NH,
      (size_t)SQ * DM, (size_t)DM * DH, (size_t)DH, (size_t)SQ * DH);
  gemm_bf16_kernel<1><<<gp, blk, 0, stream>>>(eb, wktb, bk, Kb, SKV, DH, DM, NH,
      (size_t)SKV * DM, (size_t)DM * DH, (size_t)DH, (size_t)SKV * DH);
  gemm_bf16_kernel<2><<<gp, blk, 0, stream>>>(eb, wvtb, bv, Vtb, SKV, DH, DM, NH,
      (size_t)SKV * DM, (size_t)DM * DH, (size_t)DH, (size_t)SKV * DH);

  flash_attn_kernel<<<dim3(SQ / 64, B_ * NH), blk, 0, stream>>>(Qb, Kb, Vtb, Hb);

  gemm_bf16_kernel<0><<<dim3(DM / 64, (B_ * SQ) / 64, 1), blk, 0, stream>>>(
      Hb, wotb, bo, out, B_ * SQ, DM, DM, 1, 0, 0, 0, 0);
}
